// GLMAttention_5231270167125
// MI455X (gfx1250) — compile-verified, matrix operand order checked by reading, not run on hardware
//
#include <hip/hip_runtime.h>
#include <hip/hip_bf16.h>

// GLM attention for gfx1250 (MI455X): bf16 WMMA everywhere, f32 accumulate.
// Pipeline: [gemm_wmma f32->f32 qkv+bias] -> [rope pack q/k bf16, v^T bf16]
//           -> [flash attention w/ WMMA] -> [gemm_wmma bf16 ctx x f32 w_o -> f32 out]

typedef __bf16 bf16;
typedef __attribute__((ext_vector_type(16))) __bf16 v16bf;
typedef __attribute__((ext_vector_type(8)))  __bf16 v8bf;
typedef __attribute__((ext_vector_type(8)))  float  v8f;

#define BATCH 2
#define SEQ   2048
#define HID   4096
#define NH    32
#define NKV   2
#define HD    128
#define QKVN  4608      // NH*HD + 2*NKV*HD
#define VOFF  4352      // NH*HD + NKV*HD

// Load a 16x32-bf16 A/B fragment slice: elements 0..7 contiguous at p,
// elements 8..15 contiguous at p+16 (ISA 16-bit A-matrix layout, wave32).
__device__ __forceinline__ v16bf ld_frag(const bf16* p) {
  union { v16bf v; v8bf h[2]; } u;
  u.h[0] = *(const v8bf*)p;
  u.h[1] = *(const v8bf*)(p + 16);
  return u.v;
}

__device__ __forceinline__ v8f wmma_bf16(v16bf a, v16bf b, v8f c) {
  return __builtin_amdgcn_wmma_f32_16x16x32_bf16(false, a, false, b, (short)0, c, false, false);
}

// ---------------------------------------------------------------------------
// Tiled GEMM: C[M,N] = A[M,K] * B[K,N] (+bias), A/B converted to bf16 in the
// LDS loaders, accumulate f32 via v_wmma_f32_16x16x32_bf16. Block tile 128x128,
// BK=64, 8 waves (4x2), wave tile 32x64 (2x4 C tiles).
// ---------------------------------------------------------------------------
template <typename TA, typename TB>
__global__ __launch_bounds__(256) void gemm_wmma(
    const TA* __restrict__ A, const TB* __restrict__ Bm,
    const float* __restrict__ bias, float* __restrict__ C,
    int M, int N, int K)
{
  __shared__ __align__(16) bf16 As[128][64];   // row-major A tile (16KB)
  __shared__ __align__(16) bf16 BsT[128][64];  // transposed B tile [n][k] (16KB)

  const int tid   = threadIdx.x;
  const int lane  = tid & 31;
  const int wave  = tid >> 5;
  const int lmod  = lane & 15;
  const int lhalf = lane >> 4;
  const int bm = blockIdx.y * 128;
  const int bn = blockIdx.x * 128;
  const int wm = (wave >> 1) * 32;   // 0,32,64,96
  const int wn = (wave & 1) * 64;    // 0,64

  // loader assignments
  const int arow = tid >> 1;             // 0..127
  const int acol = (tid & 1) * 32;       // 0 or 32 (32 contiguous elems)
  const int bcol = tid & 127;            // B column within tile
  const int bkof = (tid >> 7) * 32;      // 0 or 32 (32 strided rows)

  const v8f zero = {0.f,0.f,0.f,0.f,0.f,0.f,0.f,0.f};
  v8f acc[2][4];
#pragma unroll
  for (int mt = 0; mt < 2; mt++)
#pragma unroll
    for (int nt = 0; nt < 4; nt++) acc[mt][nt] = zero;

  for (int k0 = 0; k0 < K; k0 += 64) {
    const TA* ap = A + (size_t)(bm + arow) * K + k0 + acol;
#pragma unroll
    for (int j = 0; j < 32; j++) As[arow][acol + j] = (bf16)(float)ap[j];
    const TB* bp = Bm + (size_t)(k0 + bkof) * N + (bn + bcol);
#pragma unroll
    for (int j = 0; j < 32; j++) BsT[bcol][bkof + j] = (bf16)(float)bp[(size_t)j * N];
    __syncthreads();

#pragma unroll
    for (int kk = 0; kk < 2; kk++) {
      v16bf af[2], bfr[4];
#pragma unroll
      for (int mt = 0; mt < 2; mt++)
        af[mt] = ld_frag(&As[wm + mt*16 + lmod][kk*32 + lhalf*8]);
#pragma unroll
      for (int nt = 0; nt < 4; nt++)
        bfr[nt] = ld_frag(&BsT[wn + nt*16 + lmod][kk*32 + lhalf*8]);
#pragma unroll
      for (int mt = 0; mt < 2; mt++)
#pragma unroll
        for (int nt = 0; nt < 4; nt++)
          acc[mt][nt] = wmma_bf16(af[mt], bfr[nt], acc[mt][nt]);
    }
    __syncthreads();
  }

  // epilogue: C layout (VGPR r -> row r + 8*lhalf, col = lmod within tile)
#pragma unroll
  for (int mt = 0; mt < 2; mt++) {
#pragma unroll
    for (int nt = 0; nt < 4; nt++) {
      const int col  = bn + wn + nt*16 + lmod;
      const float bv = bias ? bias[col] : 0.0f;
      const int row0 = bm + wm + mt*16 + 8*lhalf;
#pragma unroll
      for (int r = 0; r < 8; r++)
        C[(size_t)(row0 + r) * N + col] = acc[mt][nt][r] + bv;
    }
  }
}

// ---------------------------------------------------------------------------
// RoPE for Q and K, fp32 qkv -> bf16 packed.  One thread per (token, head, pair).
// heads 0..31 = Q, 32..33 = K.
// ---------------------------------------------------------------------------
__global__ __launch_bounds__(256) void rope_qk_kernel(
    const float* __restrict__ qkv, const int* __restrict__ pos,
    bf16* __restrict__ q, bf16* __restrict__ k)
{
  const int idx   = blockIdx.x * blockDim.x + threadIdx.x;
  const int d     = idx & 63;
  const int head  = (idx >> 6) % 34;
  const int token = idx / (34 * 64);
  if (token >= BATCH * SEQ) return;

  const float* row = qkv + (size_t)token * QKVN;
  const float p    = (float)pos[token];
  // inv_freq = 10000^(-(2d)/128)
  const float freq = __expf(-(float)(2 * d) * (9.210340371976184f / 128.0f));
  const float ang  = p * freq;
  const float c = cosf(ang);
  const float s = sinf(ang);

  if (head < 32) {
    const float x1 = row[head * 128 + d];
    const float x2 = row[head * 128 + 64 + d];
    bf16* op = q + (size_t)token * (NH * HD) + head * 128 + d;
    op[0]  = (bf16)(x1 * c - x2 * s);
    op[64] = (bf16)(x2 * c + x1 * s);
  } else {
    const int kh = head - 32;
    const float x1 = row[NH * HD + kh * 128 + d];
    const float x2 = row[NH * HD + kh * 128 + 64 + d];
    bf16* op = k + (size_t)token * (NKV * HD) + kh * 128 + d;
    op[0]  = (bf16)(x1 * c - x2 * s);
    op[64] = (bf16)(x2 * c + x1 * s);
  }
}

// V packed transposed: vT[b][kvh][d][s] so attention P@V B-fragments are
// contiguous 16-byte loads.
__global__ __launch_bounds__(256) void pack_vT_kernel(
    const float* __restrict__ qkv, bf16* __restrict__ vT)
{
  const int idx   = blockIdx.x * blockDim.x + threadIdx.x;
  const int d     = idx & 127;
  const int kh    = (idx >> 7) & 1;
  const int token = idx >> 8;
  if (token >= BATCH * SEQ) return;
  const int b = token >> 11;      // /SEQ
  const int s = token & 2047;     // %SEQ
  const float v = qkv[(size_t)token * QKVN + VOFF + kh * 128 + d];
  vT[(((size_t)b * NKV + kh) * HD + d) * SEQ + s] = (bf16)v;
}

// ---------------------------------------------------------------------------
// Flash attention. Grid (S/64, NH, B), block = 128 (4 waves). Each wave owns
// 16 query rows; online softmax over 32-key blocks; scores and P@V via WMMA.
// ---------------------------------------------------------------------------
__global__ __launch_bounds__(128) void attn_kernel(
    const bf16* __restrict__ Q, const bf16* __restrict__ K,
    const bf16* __restrict__ Vt, bf16* __restrict__ ctx)
{
  __shared__ __align__(16) bf16 Pl[4][16 * 32];   // per-wave P staging (1KB each)

  const int lane  = threadIdx.x & 31;
  const int wave  = threadIdx.x >> 5;
  const int lmod  = lane & 15;
  const int lhalf = lane >> 4;
  const int qbase = blockIdx.x * 64 + wave * 16;
  const int h     = blockIdx.y;
  const int b     = blockIdx.z;
  const int kvh   = h >> 4;                // G = NH/NKV = 16
  const float scale = 0.08838834764831845f;   // 1/sqrt(128)

  // Q A-fragments for 16 rows x HD=128 (4 K-chunks of 32)
  v16bf qf[4];
  {
    const bf16* qp = Q + (((size_t)(b * SEQ + qbase + lmod)) * NH + h) * HD + lhalf * 8;
#pragma unroll
    for (int kk = 0; kk < 4; kk++) qf[kk] = ld_frag(qp + kk * 32);
  }

  const v8f zero = {0.f,0.f,0.f,0.f,0.f,0.f,0.f,0.f};
  v8f acc[8];
  float m[8], lsum[8];
#pragma unroll
  for (int t = 0; t < 8; t++) acc[t] = zero;
#pragma unroll
  for (int r = 0; r < 8; r++) { m[r] = -1e30f; lsum[r] = 0.f; }

  const bf16* Kbase = K + (size_t)b * SEQ * (NKV * HD) + kvh * HD;
  const bf16* Vbase = Vt + (((size_t)b * NKV + kvh) * HD) * SEQ;
  bf16* pl = &Pl[wave][0];

  const int nblocks = (qbase + 16 + 31) >> 5;
  for (int nb = 0; nb < nblocks; nb++) {
    const int n0 = nb * 32;

    // S = Q @ K^T for 32 keys: two 16x16 score tiles
    v8f s0 = zero, s1 = zero;
#pragma unroll
    for (int kk = 0; kk < 4; kk++) {
      const bf16* kp0 = Kbase + (size_t)(n0 + lmod) * (NKV * HD) + kk * 32 + lhalf * 8;
      v16bf b0 = ld_frag(kp0);
      v16bf b1 = ld_frag(kp0 + 16 * (NKV * HD));
      s0 = wmma_bf16(qf[kk], b0, s0);
      s1 = wmma_bf16(qf[kk], b1, s1);
    }

    // online softmax per row (row = r + 8*lhalf; cols striped over 16 lanes)
    float alpha[8];
#pragma unroll
    for (int r = 0; r < 8; r++) {
      const int qr  = qbase + r + 8 * lhalf;
      const int k0i = n0 + lmod;
      const int k1i = n0 + 16 + lmod;
      float a0 = (k0i <= qr) ? s0[r] * scale : -1e30f;
      float a1 = (k1i <= qr) ? s1[r] * scale : -1e30f;
      float v = fmaxf(a0, a1);
      v = fmaxf(v, __shfl_xor(v, 8));
      v = fmaxf(v, __shfl_xor(v, 4));
      v = fmaxf(v, __shfl_xor(v, 2));
      v = fmaxf(v, __shfl_xor(v, 1));
      const float mnew = fmaxf(m[r], v);
      alpha[r] = __expf(m[r] - mnew);
      const float p0 = __expf(a0 - mnew);
      const float p1 = __expf(a1 - mnew);
      float ps = p0 + p1;
      ps += __shfl_xor(ps, 8);
      ps += __shfl_xor(ps, 4);
      ps += __shfl_xor(ps, 2);
      ps += __shfl_xor(ps, 1);
      lsum[r] = lsum[r] * alpha[r] + ps;
      m[r] = mnew;
      const int row = r + 8 * lhalf;
      pl[row * 32 + lmod]      = (bf16)p0;
      pl[row * 32 + 16 + lmod] = (bf16)p1;
    }
#pragma unroll
    for (int t = 0; t < 8; t++)
#pragma unroll
      for (int r = 0; r < 8; r++) acc[t][r] *= alpha[r];

    __asm__ volatile("s_wait_dscnt 0" ::: "memory");   // P stores visible (same wave)

    // P A-fragment (16x32) from LDS, then ctx += P @ V
    const v16bf pf = ld_frag(pl + lmod * 32 + lhalf * 8);
#pragma unroll
    for (int t = 0; t < 8; t++) {
      const bf16* vp = Vbase + (size_t)(t * 16 + lmod) * SEQ + n0 + lhalf * 8;
      acc[t] = wmma_bf16(pf, ld_frag(vp), acc[t]);
    }
  }

  // normalize and write ctx (bf16, (B,S,NH*HD))
#pragma unroll
  for (int r = 0; r < 8; r++) lsum[r] = 1.0f / lsum[r];
#pragma unroll
  for (int t = 0; t < 8; t++) {
#pragma unroll
    for (int r = 0; r < 8; r++) {
      const int row = qbase + r + 8 * lhalf;
      ctx[(((size_t)(b * SEQ + row)) * NH + h) * HD + t * 16 + lmod] =
          (bf16)(acc[t][r] * lsum[r]);
    }
  }
}

// ---------------------------------------------------------------------------
extern "C" void kernel_launch(void* const* d_in, const int* in_sizes, int n_in,
                              void* d_out, int out_size, void* d_ws, size_t ws_size,
                              hipStream_t stream)
{
  const float* hidden    = (const float*)d_in[0];
  const int*   positions = (const int*)d_in[1];
  const float* w_qkv     = (const float*)d_in[2];
  const float* b_qkv     = (const float*)d_in[3];
  const float* w_o       = (const float*)d_in[4];
  float* out = (float*)d_out;

  char* ws = (char*)d_ws;
  size_t o = 0;
  float* qkv_f = (float*)(ws + o); o += (size_t)BATCH * SEQ * QKVN * 4;      // 72 MB
  bf16*  qb    = (bf16*)(ws + o);  o += (size_t)BATCH * SEQ * NH * HD * 2;   // 32 MB
  bf16*  kb    = (bf16*)(ws + o);  o += (size_t)BATCH * SEQ * NKV * HD * 2;  //  2 MB
  bf16*  vT    = (bf16*)(ws + o);  o += (size_t)BATCH * SEQ * NKV * HD * 2;  //  2 MB
  bf16*  ctx   = (bf16*)(ws + o);  o += (size_t)BATCH * SEQ * NH * HD * 2;   // 32 MB
  (void)ws_size; (void)in_sizes; (void)n_in; (void)out_size;

  const int M = BATCH * SEQ;  // 4096

  // 1) qkv = hidden @ w_qkv + b_qkv   (f32 in, bf16 WMMA, f32 out)
  {
    dim3 grid(QKVN / 128, M / 128);   // 36 x 32
    gemm_wmma<float, float><<<grid, 256, 0, stream>>>(
        hidden, w_qkv, b_qkv, qkv_f, M, QKVN, HID);
  }
  // 2) RoPE(q,k) -> bf16; v -> bf16 transposed
  {
    const int total = M * 34 * 64;
    rope_qk_kernel<<<total / 256, 256, 0, stream>>>(qkv_f, positions, qb, kb);
    const int vtotal = M * NKV * HD;
    pack_vT_kernel<<<vtotal / 256, 256, 0, stream>>>(qkv_f, vT);
  }
  // 3) flash attention -> ctx bf16
  {
    dim3 grid(SEQ / 64, NH, BATCH);   // 32 x 32 x 2
    attn_kernel<<<grid, 128, 0, stream>>>(qb, kb, vT, ctx);
  }
  // 4) out = ctx @ w_o   (bf16 A, f32 B, f32 out)
  {
    dim3 grid(HID / 128, M / 128);    // 32 x 32
    gemm_wmma<bf16, float><<<grid, 256, 0, stream>>>(
        ctx, w_o, nullptr, out, M, HID, HID);
  }
}
